// ContextGenerator_10711648436774
// MI455X (gfx1250) — matrix operand order repaired and checked by reading, not compiled, not run on hardware
//
#include <hip/hip_runtime.h>

// ---------------------------------------------------------------------------
// MI455X (gfx1250): bf16 WMMA (16x16x32, f32 accum) for all four GEMMs,
// 2x2 register tiling per wave + double-buffered K loops with all dims
// compile-time; ping-pong async LDS staging (global_load_async_to_lds_b128 +
// s_wait_asynccnt) overlapping compute in the final GEMM; f32 softmax in
// place in the probs output region.
// ---------------------------------------------------------------------------

#define DEV __device__ __forceinline__

typedef __bf16 bf16_t;
typedef __attribute__((ext_vector_type(16))) __bf16 v16bf;
typedef __attribute__((ext_vector_type(8)))  __bf16 v8bf;
typedef __attribute__((ext_vector_type(8)))  float  v8f;

// Problem constants
constexpr int Bn = 64;      // batch
constexpr int Tn = 1024;    // context length
constexpr int Fn = 384;     // feature dim 256+96+32
constexpr int Un = 256;     // UNITS
constexpr int Mn = 1038;    // MEM_TOTAL = 1024 + 14
constexpr int MP = 1056;    // M padded to multiple of 32 (33*32)
constexpr int DSs = 256, DAs = 96, DRs = 32;

// Workspace layout (bytes). All offsets are multiples of 256.
constexpr size_t SZ_MEM  = (size_t)Bn * Mn * Fn * 2;   // bf16 [B][M][F]
constexpr size_t SZ_MEMT = (size_t)Bn * Fn * MP * 2;   // bf16 [B][F][MP]
constexpr size_t SZ_QK   = (size_t)Bn * Mn * Un * 2;   // bf16 [B][M][U]
constexpr size_t SZ_WT   = (size_t)Un * Fn * 2;        // bf16 [U][F]
constexpr size_t SZ_KER  = (size_t)Mn * MP * 2;        // bf16 [M][MP]

constexpr size_t OFF_MEM  = 0;
constexpr size_t OFF_MEMT = OFF_MEM  + SZ_MEM;
constexpr size_t OFF_Q    = OFF_MEMT + SZ_MEMT;
constexpr size_t OFF_K    = OFF_Q    + SZ_QK;
constexpr size_t OFF_ATTT = OFF_K    + SZ_QK;
constexpr size_t OFF_WQT  = OFF_ATTT + SZ_MEMT;
constexpr size_t OFF_WKT  = OFF_WQT  + SZ_WT;
constexpr size_t OFF_KER  = OFF_WKT  + SZ_WT;

DEV int imin(int a, int b) { return a < b ? a : b; }

DEV v16bf make16(v8bf lo, v8bf hi) {
  union { v16bf v; v8bf h[2]; } u;
  u.h[0] = lo; u.h[1] = hi;
  return u.v;
}

// WMMA A operand (16x32 bf16), row-major, 16B-aligned row stride.
// lane L: row = m0 + (L&15); K chunks at k0+(L>>4)*8 and +16.
DEV v16bf load_a_bf16(const bf16_t* __restrict__ A, int lda, int m0, int k0,
                      int mmax, int lane) {
  int r = lane & 15, hi = lane >> 4;
  int row = imin(m0 + r, mmax - 1);
  const bf16_t* p = A + (size_t)row * lda + k0 + hi * 8;
  return make16(*(const v8bf*)p, *(const v8bf*)(p + 16));
}

// WMMA B operand (32x16 bf16) from a K-contiguous ("BT") layout:
// BT[n*ldb + k]. lane L: col n = n0+(L&15); K chunks at k0+(L>>4)*8 and +16.
DEV v16bf load_bt_bf16(const bf16_t* __restrict__ BT, int ldb, int n0, int k0,
                       int nmax, int lane) {
  int c = imin(n0 + (lane & 15), nmax - 1);
  int kk = k0 + (lane >> 4) * 8;
  const bf16_t* p = BT + (size_t)c * ldb + kk;
  return make16(*(const v8bf*)p, *(const v8bf*)(p + 16));
}

DEV v8f wmma_bf16(v16bf a, v16bf b, v8f c) {
  return __builtin_amdgcn_wmma_f32_16x16x32_bf16(false, a, false, b,
                                                 (short)0, c, false, false);
}

// ---------------------------------------------------------------------------
// Stage 0: zero padded buffers (b128 stores)
// ---------------------------------------------------------------------------
__global__ void zero_kernel(uint4* __restrict__ p, size_t n16) {
  size_t i = (size_t)blockIdx.x * blockDim.x + threadIdx.x;
  size_t stride = (size_t)gridDim.x * blockDim.x;
  uint4 z; z.x = z.y = z.z = z.w = 0u;
  for (; i < n16; i += stride) p[i] = z;
}

// ---------------------------------------------------------------------------
// Stage 1: build helix memory (compressed rows + recent), bf16 + transposed
// ---------------------------------------------------------------------------
DEV float ctx_at(const float* __restrict__ xs, const float* __restrict__ xa,
                 const float* __restrict__ xr, int b, int t, int f) {
  if (f < DSs) return xs[((size_t)b * Tn + t) * DSs + f];
  if (f < DSs + DAs) return xa[((size_t)b * Tn + t) * DAs + (f - DSs)];
  return xr[((size_t)b * Tn + t) * DRs + (f - DSs - DAs)];
}

__global__ void build_mem_kernel(const float* __restrict__ xs,
                                 const float* __restrict__ xa,
                                 const float* __restrict__ xr,
                                 const float* __restrict__ w1,
                                 const float* __restrict__ w2,
                                 const float* __restrict__ w3,
                                 bf16_t* __restrict__ mem,
                                 bf16_t* __restrict__ memT) {
  size_t idx = (size_t)blockIdx.x * 256 + threadIdx.x;
  if (idx >= (size_t)Bn * Mn * Fn) return;
  int f = (int)(idx % Fn);
  int m = (int)((idx / Fn) % Mn);
  int b = (int)(idx / ((size_t)Fn * Mn));
  float v;
  if (m < 14) {
    int L, si; const float* w;
    if (m < 8)       { L = 128; si = m;      w = w3; }
    else if (m < 12) { L = 256; si = m - 8;  w = w2; }
    else             { L = 512; si = m - 12; w = w1; }
    int t0 = si * L;
    float acc = 0.f;
    for (int l = 0; l < L; ++l)
      acc = fmaf(ctx_at(xs, xa, xr, b, t0 + l, f), w[l], acc);
    v = acc;
  } else {
    v = ctx_at(xs, xa, xr, b, m - 14, f);
  }
  mem[((size_t)b * Mn + m) * Fn + f]  = (bf16_t)v;
  memT[((size_t)b * Fn + f) * MP + m] = (bf16_t)v;
}

// Wq/Wk [F][U] f32 -> transposed bf16 [U][F]
__global__ void wt_kernel(const float* __restrict__ W, bf16_t* __restrict__ WT) {
  int idx = blockIdx.x * 256 + threadIdx.x;
  if (idx >= Un * Fn) return;
  int u = idx / Fn, f = idx % Fn;
  WT[(size_t)u * Fn + f] = (bf16_t)W[(size_t)f * Un + u];
}

// kernel [M][M] f32 -> bf16 [M][MP] (padding pre-zeroed)
__global__ void ker_kernel(const float* __restrict__ Kin, bf16_t* __restrict__ Kw) {
  size_t idx = (size_t)blockIdx.x * 256 + threadIdx.x;
  if (idx >= (size_t)Mn * Mn) return;
  int n = (int)(idx % Mn), m = (int)(idx / Mn);
  Kw[(size_t)m * MP + n] = (bf16_t)Kin[idx];
}

// ---------------------------------------------------------------------------
// GEMM 1: q/k = mem @ W + b. 2x2 WMMA tiles/wave (32x32), double-buffered K.
// block = 8 waves as 4(m) x 2(n) -> block tile 128 x 64. All dims constexpr:
// A [Mn x Fn] lda=Fn, BT = W^T [Un x Fn] ldb=Fn, C [Mn x Un] ldc=Un, K=Fn.
// ---------------------------------------------------------------------------
__global__ void gemm_bias_kernel(const bf16_t* __restrict__ A,
                                 const bf16_t* __restrict__ BT,
                                 const float* __restrict__ bias,
                                 bf16_t* __restrict__ C) {
  int b = blockIdx.z;
  int wave = threadIdx.x >> 5, lane = threadIdx.x & 31;
  int m0 = blockIdx.x * 128 + (wave & 3) * 32;
  int n0 = blockIdx.y * 64 + (wave >> 2) * 32;
  const bf16_t* Ab = A + (size_t)b * (Mn * Fn);
  v16bf a0c = load_a_bf16(Ab, Fn, m0,      0, Mn, lane);
  v16bf a1c = load_a_bf16(Ab, Fn, m0 + 16, 0, Mn, lane);
  v16bf b0c = load_bt_bf16(BT, Fn, n0,      0, Un, lane);
  v16bf b1c = load_bt_bf16(BT, Fn, n0 + 16, 0, Un, lane);
  v8f acc00 = {}, acc01 = {}, acc10 = {}, acc11 = {};
#pragma unroll 4
  for (int k0 = 0; k0 < Fn; k0 += 32) {
    int kn = (k0 + 32 < Fn) ? (k0 + 32) : k0;   // folds at compile time
    v16bf a0n = load_a_bf16(Ab, Fn, m0,      kn, Mn, lane);
    v16bf a1n = load_a_bf16(Ab, Fn, m0 + 16, kn, Mn, lane);
    v16bf b0n = load_bt_bf16(BT, Fn, n0,      kn, Un, lane);
    v16bf b1n = load_bt_bf16(BT, Fn, n0 + 16, kn, Un, lane);
    acc00 = wmma_bf16(a0c, b0c, acc00);
    acc01 = wmma_bf16(a0c, b1c, acc01);
    acc10 = wmma_bf16(a1c, b0c, acc10);
    acc11 = wmma_bf16(a1c, b1c, acc11);
    a0c = a0n; a1c = a1n; b0c = b0n; b1c = b1n;
  }
  int hi = lane >> 4;
  bf16_t* Cb = C + (size_t)b * (Mn * Un);
  const v8f accs[4] = {acc00, acc01, acc10, acc11};
#pragma unroll
  for (int t = 0; t < 4; ++t) {
    int mb = m0 + (t >> 1) * 16, c = n0 + (t & 1) * 16 + (lane & 15);
    float bs = bias[c];
#pragma unroll
    for (int r = 0; r < 8; ++r) {
      int m = mb + hi * 8 + r;
      if (m < Mn) Cb[(size_t)m * Un + c] = (bf16_t)(accs[t][r] + bs);
    }
  }
}

// ---------------------------------------------------------------------------
// GEMM 2: scores = (q @ k^T) / 16 -> f32 into probs output region.
// A = q [Mn x Un] lda=Un; BT = k [Mn x Un] ldb=Un; C [Mn x Mn]; K=Un.
// ---------------------------------------------------------------------------
__global__ void gemm_scores_kernel(const bf16_t* __restrict__ A,
                                   const bf16_t* __restrict__ BT,
                                   float* __restrict__ C) {
  int b = blockIdx.z;
  int wave = threadIdx.x >> 5, lane = threadIdx.x & 31;
  int m0 = blockIdx.x * 128 + (wave & 3) * 32;
  int n0 = blockIdx.y * 64 + (wave >> 2) * 32;
  const bf16_t* Ab  = A  + (size_t)b * (Mn * Un);
  const bf16_t* BTb = BT + (size_t)b * (Mn * Un);
  v16bf a0c = load_a_bf16(Ab, Un, m0,      0, Mn, lane);
  v16bf a1c = load_a_bf16(Ab, Un, m0 + 16, 0, Mn, lane);
  v16bf b0c = load_bt_bf16(BTb, Un, n0,      0, Mn, lane);
  v16bf b1c = load_bt_bf16(BTb, Un, n0 + 16, 0, Mn, lane);
  v8f acc00 = {}, acc01 = {}, acc10 = {}, acc11 = {};
#pragma unroll 4
  for (int k0 = 0; k0 < Un; k0 += 32) {
    int kn = (k0 + 32 < Un) ? (k0 + 32) : k0;   // folds at compile time
    v16bf a0n = load_a_bf16(Ab, Un, m0,      kn, Mn, lane);
    v16bf a1n = load_a_bf16(Ab, Un, m0 + 16, kn, Mn, lane);
    v16bf b0n = load_bt_bf16(BTb, Un, n0,      kn, Mn, lane);
    v16bf b1n = load_bt_bf16(BTb, Un, n0 + 16, kn, Mn, lane);
    acc00 = wmma_bf16(a0c, b0c, acc00);
    acc01 = wmma_bf16(a0c, b1c, acc01);
    acc10 = wmma_bf16(a1c, b0c, acc10);
    acc11 = wmma_bf16(a1c, b1c, acc11);
    a0c = a0n; a1c = a1n; b0c = b0n; b1c = b1n;
  }
  int hi = lane >> 4;
  float* Cb = C + (size_t)b * ((size_t)Mn * Mn);
  constexpr float scale = 1.0f / 16.0f;   // 1/sqrt(UNITS)
  const v8f accs[4] = {acc00, acc01, acc10, acc11};
#pragma unroll
  for (int t = 0; t < 4; ++t) {
    int mb = m0 + (t >> 1) * 16, c = n0 + (t & 1) * 16 + (lane & 15);
#pragma unroll
    for (int r = 0; r < 8; ++r) {
      int m = mb + hi * 8 + r;
      if (m < Mn && c < Mn) Cb[(size_t)m * Mn + c] = accs[t][r] * scale;
    }
  }
}

// ---------------------------------------------------------------------------
// Softmax over each row of probs [B*Mn rows, Mn cols] in place (f32)
// ---------------------------------------------------------------------------
__global__ void softmax_kernel(float* __restrict__ probs) {
  __shared__ float red[256];
  float* p = probs + (size_t)blockIdx.x * Mn;
  int tid = threadIdx.x;
  float mx = -3.4e38f;
  for (int i = tid; i < Mn; i += 256) mx = fmaxf(mx, p[i]);
  red[tid] = mx; __syncthreads();
  for (int s = 128; s > 0; s >>= 1) {
    if (tid < s) red[tid] = fmaxf(red[tid], red[tid + s]);
    __syncthreads();
  }
  mx = red[0]; __syncthreads();
  float sum = 0.f;
  for (int i = tid; i < Mn; i += 256) { float e = __expf(p[i] - mx); p[i] = e; sum += e; }
  red[tid] = sum; __syncthreads();
  for (int s = 128; s > 0; s >>= 1) {
    if (tid < s) red[tid] += red[tid + s];
    __syncthreads();
  }
  float inv = 1.0f / red[0];
  for (int i = tid; i < Mn; i += 256) p[i] *= inv;
}

// ---------------------------------------------------------------------------
// GEMM 3: att = probs @ mem. A = probs f32 (bf16-converted in-register,
// zero-masked K tail at Mn), BT = memT [Fn x MP] (zero-padded). 2x2 tiles.
// C stored TRANSPOSED bf16 into attT[B][Fn][MP].
// ---------------------------------------------------------------------------
DEV v16bf load_a_f32_bf16(const float* __restrict__ row, int kk) {
  union { v16bf v; bf16_t e[16]; } ua;
#pragma unroll
  for (int j = 0; j < 8; ++j) {
    int kA = kk + j;
    ua.e[j] = (kA < Mn) ? (bf16_t)row[kA] : (bf16_t)0.0f;
  }
#pragma unroll
  for (int j = 0; j < 8; ++j) {
    int kA = kk + 16 + j;
    ua.e[8 + j] = (kA < Mn) ? (bf16_t)row[kA] : (bf16_t)0.0f;
  }
  return ua.v;
}

__global__ void gemm_pv_kernel(const float* __restrict__ P,
                               const bf16_t* __restrict__ BT,
                               bf16_t* __restrict__ CT) {
  int b = blockIdx.z;
  int wave = threadIdx.x >> 5, lane = threadIdx.x & 31;
  int m0 = blockIdx.x * 128 + (wave & 3) * 32;
  int n0 = blockIdx.y * 64 + (wave >> 2) * 32;
  const float*  Pb  = P  + (size_t)b * ((size_t)Mn * Mn);
  const bf16_t* BTb = BT + (size_t)b * (Fn * MP);
  int r = lane & 15, hi = lane >> 4;
  const float* ap0 = Pb + (size_t)imin(m0 + r,      Mn - 1) * Mn;
  const float* ap1 = Pb + (size_t)imin(m0 + 16 + r, Mn - 1) * Mn;
  v8f acc00 = {}, acc01 = {}, acc10 = {}, acc11 = {};
  for (int k0 = 0; k0 < MP; k0 += 32) {
    int kk = k0 + hi * 8;
    if (k0 + 32 < MP) __builtin_prefetch(ap0 + kk + 32, 0, 1);
    v16bf a0 = load_a_f32_bf16(ap0, kk);
    v16bf a1 = load_a_f32_bf16(ap1, kk);
    v16bf b0 = load_bt_bf16(BTb, MP, n0,      k0, Fn, lane);
    v16bf b1 = load_bt_bf16(BTb, MP, n0 + 16, k0, Fn, lane);
    acc00 = wmma_bf16(a0, b0, acc00);
    acc01 = wmma_bf16(a0, b1, acc01);
    acc10 = wmma_bf16(a1, b0, acc10);
    acc11 = wmma_bf16(a1, b1, acc11);
  }
  bf16_t* CTb = CT + (size_t)b * (Fn * MP);
  const v8f accs[4] = {acc00, acc01, acc10, acc11};
#pragma unroll
  for (int t = 0; t < 4; ++t) {
    int mb = m0 + (t >> 1) * 16, c = n0 + (t & 1) * 16 + (lane & 15);
#pragma unroll
    for (int rr = 0; rr < 8; ++rr) {
      int m = mb + hi * 8 + rr;
      if (m < Mn) CTb[(size_t)c * MP + m] = (bf16_t)accs[t][rr];
    }
  }
}

// ---------------------------------------------------------------------------
// GEMM 4: out = kernel @ att. A (kernel bf16 [Mn][MP], batch-invariant) is
// ping-pong staged per K-slice into LDS with gfx1250 async loads
// (global_load_async_to_lds_b128 + s_wait_asynccnt): the stage of slice i+1
// overlaps the WMMAs of slice i, one barrier per iteration. A operands are
// read from LDS (ds_load_b128); B = attT; C f32 -> d_out region 1.
// ---------------------------------------------------------------------------
__global__ void gemm_out_kernel(const bf16_t* __restrict__ A,
                                const bf16_t* __restrict__ BT,
                                float* __restrict__ C) {
  __shared__ __align__(16) bf16_t As[2][128 * 32];   // 2 x 8 KB ping-pong
  int b = blockIdx.z;
  int tid = threadIdx.x;
  int wave = tid >> 5, lane = tid & 31;
  int m0b = blockIdx.x * 128;
  int mloc = (wave & 3) * 32;
  int n0 = blockIdx.y * 64 + (wave >> 2) * 32;
  const bf16_t* BTb = BT + (size_t)b * (Fn * MP);

  // two 16B chunks per thread per K-slice: 512 chunks = 128 rows x 4 chunks
  int row0 = tid >> 2,         ch0 = tid & 3;
  int row1 = (tid + 256) >> 2, ch1 = tid & 3;
  const bf16_t* g0base = A + (size_t)imin(m0b + row0, Mn - 1) * MP + ch0 * 8;
  const bf16_t* g1base = A + (size_t)imin(m0b + row1, Mn - 1) * MP + ch1 * 8;
  unsigned l0 = (unsigned)(uintptr_t)(&As[0][0] + row0 * 32 + ch0 * 8);
  unsigned l1 = (unsigned)(uintptr_t)(&As[0][0] + row1 * 32 + ch1 * 8);

  int r = lane & 15, hi = lane >> 4;
  const bf16_t* ap0 = &As[0][0] + (mloc + r) * 32 + hi * 8;
  const bf16_t* ap1 = &As[0][0] + (mloc + 16 + r) * 32 + hi * 8;

  // prologue: stage slice 0 into buffer 0
  asm volatile("global_load_async_to_lds_b128 %0, %1, off"
               :: "v"(l0), "v"(g0base) : "memory");
  asm volatile("global_load_async_to_lds_b128 %0, %1, off"
               :: "v"(l1), "v"(g1base) : "memory");

  v8f acc00 = {}, acc01 = {}, acc10 = {}, acc11 = {};
  int cur = 0;
  for (int k0 = 0; k0 < MP; k0 += 32) {
    asm volatile("s_wait_asynccnt 0x0" ::: "memory");   // current slice landed
    __syncthreads();                                    // ... for all waves
    // stage next slice into the other buffer (overlaps compute below)
    int nxt = cur ^ 1;
    int kn = (k0 + 32 < MP) ? (k0 + 32) : k0;
    unsigned boff = (unsigned)(nxt * 128 * 32 * sizeof(bf16_t));
    asm volatile("global_load_async_to_lds_b128 %0, %1, off"
                 :: "v"(l0 + boff), "v"(g0base + kn) : "memory");
    asm volatile("global_load_async_to_lds_b128 %0, %1, off"
                 :: "v"(l1 + boff), "v"(g1base + kn) : "memory");

    const bf16_t* a0p = ap0 + cur * (128 * 32);
    const bf16_t* a1p = ap1 + cur * (128 * 32);
    v16bf a0 = make16(*(const v8bf*)a0p, *(const v8bf*)(a0p + 16));
    v16bf a1 = make16(*(const v8bf*)a1p, *(const v8bf*)(a1p + 16));
    v16bf b0 = load_bt_bf16(BTb, MP, n0,      k0, Fn, lane);
    v16bf b1 = load_bt_bf16(BTb, MP, n0 + 16, k0, Fn, lane);
    acc00 = wmma_bf16(a0, b0, acc00);
    acc01 = wmma_bf16(a0, b1, acc01);
    acc10 = wmma_bf16(a1, b0, acc10);
    acc11 = wmma_bf16(a1, b1, acc11);
    cur = nxt;
  }
  asm volatile("s_wait_asynccnt 0x0" ::: "memory");     // drain before exit

  float* Cb = C + (size_t)b * (Mn * Fn);
  const v8f accs[4] = {acc00, acc01, acc10, acc11};
#pragma unroll
  for (int t = 0; t < 4; ++t) {
    int mb = m0b + mloc + (t >> 1) * 16, c = n0 + (t & 1) * 16 + (lane & 15);
#pragma unroll
    for (int rr = 0; rr < 8; ++rr) {
      int m = mb + hi * 8 + rr;
      if (m < Mn) Cb[(size_t)m * Fn + c] = accs[t][rr];
    }
  }
}

// ---------------------------------------------------------------------------
extern "C" void kernel_launch(void* const* d_in, const int* in_sizes, int n_in,
                              void* d_out, int out_size, void* d_ws, size_t ws_size,
                              hipStream_t stream) {
  const float* xs   = (const float*)d_in[0];
  const float* xa   = (const float*)d_in[1];
  const float* xr   = (const float*)d_in[2];
  const float* w1   = (const float*)d_in[3];
  const float* w2   = (const float*)d_in[4];
  const float* w3   = (const float*)d_in[5];
  const float* Wq   = (const float*)d_in[6];
  const float* bq   = (const float*)d_in[7];
  const float* Wk   = (const float*)d_in[8];
  const float* bk   = (const float*)d_in[9];
  const float* Kmat = (const float*)d_in[10];

  char* ws = (char*)d_ws;
  bf16_t* mem  = (bf16_t*)(ws + OFF_MEM);
  bf16_t* memT = (bf16_t*)(ws + OFF_MEMT);
  bf16_t* qb   = (bf16_t*)(ws + OFF_Q);
  bf16_t* kb   = (bf16_t*)(ws + OFF_K);
  bf16_t* attT = (bf16_t*)(ws + OFF_ATTT);
  bf16_t* WqT  = (bf16_t*)(ws + OFF_WQT);
  bf16_t* WkT  = (bf16_t*)(ws + OFF_WKT);
  bf16_t* kerW = (bf16_t*)(ws + OFF_KER);

  float* outp  = (float*)d_out;                         // [B][Mn][Fn] f32
  float* probs = outp + (size_t)Bn * Mn * Fn;           // [B][Mn][Mn] f32

  dim3 blk(256);

  // zero-init padded buffers (memT / attT / kernel tails must be 0)
  zero_kernel<<<2048, blk, 0, stream>>>((uint4*)memT, SZ_MEMT / 16);
  zero_kernel<<<2048, blk, 0, stream>>>((uint4*)attT, SZ_MEMT / 16);
  zero_kernel<<<512,  blk, 0, stream>>>((uint4*)kerW, SZ_KER  / 16);

  // build compressive memory + bf16 weight copies
  build_mem_kernel<<<(unsigned)(((size_t)Bn * Mn * Fn + 255) / 256), blk, 0, stream>>>(
      xs, xa, xr, w1, w2, w3, mem, memT);
  wt_kernel<<<(Un * Fn + 255) / 256, blk, 0, stream>>>(Wq, WqT);
  wt_kernel<<<(Un * Fn + 255) / 256, blk, 0, stream>>>(Wk, WkT);
  ker_kernel<<<(unsigned)(((size_t)Mn * Mn + 255) / 256), blk, 0, stream>>>(Kmat, kerW);

  const int MtB = (Mn + 127) / 128;                     // 9 block-rows

  // q, k projections: [1038 x 384] @ [384 x 256]
  gemm_bias_kernel<<<dim3(MtB, Un / 64, Bn), blk, 0, stream>>>(mem, WqT, bq, qb);
  gemm_bias_kernel<<<dim3(MtB, Un / 64, Bn), blk, 0, stream>>>(mem, WkT, bk, kb);

  // scores = q @ k^T / 16 -> probs region (f32)
  gemm_scores_kernel<<<dim3(MtB, (Mn + 63) / 64, Bn), blk, 0, stream>>>(qb, kb, probs);

  // softmax in place
  softmax_kernel<<<Bn * Mn, blk, 0, stream>>>(probs);

  // att = probs @ mem, stored transposed bf16
  gemm_pv_kernel<<<dim3(MtB, Fn / 64, Bn), blk, 0, stream>>>(probs, memT, attT);

  // out = kernel @ att -> d_out region 1 (f32)
  gemm_out_kernel<<<dim3(MtB, Fn / 64, Bn), blk, 0, stream>>>(kerW, attT, outp);
}
